// Head_8839042695394
// MI455X (gfx1250) — compile-verified, matrix operand order checked by reading, not run on hardware
//
#include <hip/hip_runtime.h>
#include <hip/hip_bf16.h>

// Problem constants (match reference)
#define BB 4
#define TT 2048
#define CC 1024
#define HH 64
#define BT (BB * TT)   // 8192

typedef __attribute__((ext_vector_type(16))) __bf16 v16bf;
typedef __attribute__((ext_vector_type(8)))  __bf16 v8bf;
typedef __attribute__((ext_vector_type(4)))  __bf16 v4bf;
typedef __attribute__((ext_vector_type(8)))  float  v8f;

// ---------------------------------------------------------------------------
// WMMA v_wmma_f32_16x16x32_bf16 fragment layouts (wave32):
// A (16x32): lane -> row = lane&15, kb = (lane>>4)*8; elems 0..7 -> k=kb+i,
//            elems 8..15 -> k=kb+16+(i-8).  => two contiguous 8-elem runs.
// B (32x16): lane -> col = lane&15, same k pattern.
// C/D (16x16 f32): lane -> col = lane&15, elem v -> row = (lane>>4)*8 + v.
// ---------------------------------------------------------------------------
__device__ __forceinline__ v16bf load_frag_bf16(const __bf16* __restrict__ base) {
    // base points at the lane's first run; second run is +16 elements.
    v8bf lo = *(const v8bf*)(base);
    v8bf hi = *(const v8bf*)(base + 16);
    return __builtin_shufflevector(lo, hi, 0, 1, 2, 3, 4, 5, 6, 7,
                                           8, 9, 10, 11, 12, 13, 14, 15);
}

// ---------------------------------------------------------------------------
// Kernel 0a: x (f32) -> xb (bf16), elementwise, float4 granularity
// ---------------------------------------------------------------------------
__global__ __launch_bounds__(256) void cvt_x_kernel(
    const float* __restrict__ x, __bf16* __restrict__ xb)
{
    const size_t i = ((size_t)blockIdx.x * blockDim.x + threadIdx.x) * 4;
    float4 f = *(const float4*)(x + i);
    v4bf b = { (__bf16)f.x, (__bf16)f.y, (__bf16)f.z, (__bf16)f.w };
    *(v4bf*)(xb + i) = b;
}

// ---------------------------------------------------------------------------
// Kernel 0b: W[c][n] (f32) -> Wt[w][n][c] (bf16), transposed so the GEMM
// contraction dim (c) is contiguous for B-fragment b128 loads.
// ---------------------------------------------------------------------------
__global__ __launch_bounds__(256) void cvt_w_kernel(
    const float* __restrict__ Wq, const float* __restrict__ Wk,
    const float* __restrict__ Wv, __bf16* __restrict__ wt)
{
    const int id = blockIdx.x * blockDim.x + threadIdx.x;  // [0, 3*HH*CC)
    const int c = id & (CC - 1);
    const int n = (id >> 10) & (HH - 1);
    const int w = id >> 16;
    const float* __restrict__ W = (w == 0) ? Wq : (w == 1) ? Wk : Wv;
    wt[id] = (__bf16)W[c * HH + n];
}

// ---------------------------------------------------------------------------
// Kernel 1: projections (bf16 WMMA, f32 accumulate).
//   which 0: Q (scaled by 1/sqrt(H)) -> qo[m][h]   bf16 row-major
//   which 1: K                       -> ko[m][h]   bf16 row-major
//   which 2: V                       -> vt[b][h][t] bf16 (transposed)
// grid = (BT/16, 3), block = 128 (wave w owns N-tile w of H=64)
// ---------------------------------------------------------------------------
__global__ __launch_bounds__(128) void proj_kernel(
    const __bf16* __restrict__ xb,
    const __bf16* __restrict__ wt,
    __bf16* __restrict__ qo,
    __bf16* __restrict__ ko,
    __bf16* __restrict__ vt)
{
    const int lane  = threadIdx.x & 31;
    const int wave  = threadIdx.x >> 5;
    const int mtile = blockIdx.x;
    const int which = blockIdx.y;

    const int kb8  = (lane >> 4) << 3;
    const int arow = mtile * 16 + (lane & 15);
    const int ncol = wave * 16 + (lane & 15);

    const __bf16* __restrict__ aptr = xb + (size_t)arow * CC + kb8;
    const __bf16* __restrict__ bptr = wt + ((size_t)which * HH + ncol) * CC + kb8;

    v8f acc = {};
    for (int kk = 0; kk < CC; kk += 32) {
        v16bf a = load_frag_bf16(aptr + kk);
        v16bf b = load_frag_bf16(bptr + kk);
        acc = __builtin_amdgcn_wmma_f32_16x16x32_bf16(
                  false, a, false, b, (short)0, acc, false, false);
    }

    const int mbase = (lane >> 4) << 3;
    if (which == 2) {
        // V transposed: vt[(batch*HH + ncol)*TT + t], t contiguous over vv
        const int m0    = mtile * 16 + mbase;
        const int batch = m0 >> 11;            // /TT
        const int t0    = m0 & (TT - 1);
        v8bf o;
        #pragma unroll
        for (int vv = 0; vv < 8; ++vv) o[vv] = (__bf16)acc[vv];
        *(v8bf*)(vt + ((size_t)batch * HH + ncol) * TT + t0) = o;
    } else {
        __bf16* __restrict__ dst = (which == 0) ? qo : ko;
        const float scale = (which == 0) ? 0.125f : 1.0f;   // 1/sqrt(64)
        #pragma unroll
        for (int vv = 0; vv < 8; ++vv) {
            const int m = mtile * 16 + mbase + vv;
            dst[(size_t)m * HH + ncol] = (__bf16)(acc[vv] * scale);
        }
    }
}

// ---------------------------------------------------------------------------
// Kernel 2: causal flash attention, online softmax, 32-key blocks.
// grid = (TT/64, BB), block = 128; wave w owns query tile blockIdx.x*4+w.
// ---------------------------------------------------------------------------
__global__ __launch_bounds__(128) void attn_kernel(
    const __bf16* __restrict__ q,
    const __bf16* __restrict__ k,
    const __bf16* __restrict__ vt,
    float* __restrict__ out)
{
    __shared__ float pbuf[4][16 * 32];    // per-wave P staging (2 KB each)

    const int lane  = threadIdx.x & 31;
    const int wave  = threadIdx.x >> 5;
    const int batch = blockIdx.y;
    const int qrow0 = (blockIdx.x * 4 + wave) * 16;

    const __bf16* __restrict__ qb  = q  + (size_t)batch * TT * HH;
    const __bf16* __restrict__ kbp = k  + (size_t)batch * TT * HH;
    const __bf16* __restrict__ vtp = vt + (size_t)batch * HH * TT;

    const int ncol  = lane & 15;
    const int kb8   = (lane >> 4) << 3;
    const int mbase = kb8;
    float* __restrict__ pl = &pbuf[wave][0];

    // Q tile as two A fragments (H=64 -> 2 k-steps); scale folded into proj.
    v16bf qa0 = load_frag_bf16(qb + (size_t)(qrow0 + ncol) * HH + kb8);
    v16bf qa1 = load_frag_bf16(qb + (size_t)(qrow0 + ncol) * HH + 32 + kb8);

    v8f   o[4] = {};
    float mrow[8], lrow[8];
    #pragma unroll
    for (int vv = 0; vv < 8; ++vv) { mrow[vv] = -1e30f; lrow[vv] = 0.0f; }

    const int nblocks = qrow0 / 32 + 1;      // causal extent
    for (int j = 0; j < nblocks; ++j) {
        const int key0 = j * 32;

        if (j + 1 < nblocks) {               // prefetch next key block
            __builtin_prefetch(kbp + (size_t)(key0 + 32 + ncol) * HH, 0, 1);
            __builtin_prefetch(vtp + (size_t)ncol * TT + key0 + 32, 0, 1);
        }

        // ---- S = Q K^T : two 16-key tiles -------------------------------
        v8f s[2];
        #pragma unroll
        for (int t = 0; t < 2; ++t) {
            const int key = key0 + t * 16 + ncol;   // B-frag column
            v16bf kf0 = load_frag_bf16(kbp + (size_t)key * HH + kb8);
            v16bf kf1 = load_frag_bf16(kbp + (size_t)key * HH + 32 + kb8);
            v8f z = {};
            z = __builtin_amdgcn_wmma_f32_16x16x32_bf16(
                    false, qa0, false, kf0, (short)0, z, false, false);
            s[t] = __builtin_amdgcn_wmma_f32_16x16x32_bf16(
                    false, qa1, false, kf1, (short)0, z, false, false);
        }

        // ---- causal mask + online softmax -------------------------------
        #pragma unroll
        for (int vv = 0; vv < 8; ++vv) {
            const int qi = qrow0 + mbase + vv;
            float s0 = ((key0 + ncol)      > qi) ? -1e30f : s[0][vv];
            float s1 = ((key0 + 16 + ncol) > qi) ? -1e30f : s[1][vv];

            float mx = fmaxf(s0, s1);
            #pragma unroll
            for (int d = 1; d < 16; d <<= 1)
                mx = fmaxf(mx, __shfl_xor(mx, d, 32));

            const float newm  = fmaxf(mrow[vv], mx);
            const float alpha = __expf(mrow[vv] - newm);
            const float p0 = __expf(s0 - newm);
            const float p1 = __expf(s1 - newm);

            float rs = p0 + p1;
            #pragma unroll
            for (int d = 1; d < 16; d <<= 1)
                rs += __shfl_xor(rs, d, 32);

            lrow[vv] = lrow[vv] * alpha + rs;
            mrow[vv] = newm;
            s[0][vv] = p0;
            s[1][vv] = p1;
            #pragma unroll
            for (int n = 0; n < 4; ++n) o[n][vv] *= alpha;
        }

        // ---- P: C/D frag -> LDS (f32) -> A frag (bf16) ------------------
        #pragma unroll
        for (int t = 0; t < 2; ++t)
            #pragma unroll
            for (int vv = 0; vv < 8; ++vv)
                pl[(mbase + vv) * 32 + t * 16 + ncol] = s[t][vv];
        // same-wave DS ops are in-order; compiler inserts s_wait_dscnt

        const float* __restrict__ prow = pl + ncol * 32 + kb8;
        float4 p0 = *(const float4*)(prow);
        float4 p1 = *(const float4*)(prow + 4);
        float4 p2 = *(const float4*)(prow + 16);
        float4 p3 = *(const float4*)(prow + 20);
        v16bf pa = { (__bf16)p0.x, (__bf16)p0.y, (__bf16)p0.z, (__bf16)p0.w,
                     (__bf16)p1.x, (__bf16)p1.y, (__bf16)p1.z, (__bf16)p1.w,
                     (__bf16)p2.x, (__bf16)p2.y, (__bf16)p2.z, (__bf16)p2.w,
                     (__bf16)p3.x, (__bf16)p3.y, (__bf16)p3.z, (__bf16)p3.w };

        // ---- O += P V : K=32 keys, 4 N-tiles over H ---------------------
        #pragma unroll
        for (int n = 0; n < 4; ++n) {
            const int h = n * 16 + ncol;     // B-frag column
            v16bf vf = load_frag_bf16(vtp + (size_t)h * TT + key0 + kb8);
            o[n] = __builtin_amdgcn_wmma_f32_16x16x32_bf16(
                       false, pa, false, vf, (short)0, o[n], false, false);
        }
    }

    // ---- epilogue: normalize and store (f32 out) ------------------------
    #pragma unroll
    for (int vv = 0; vv < 8; ++vv) {
        const float inv = 1.0f / lrow[vv];
        const int   qi  = qrow0 + mbase + vv;
        #pragma unroll
        for (int n = 0; n < 4; ++n)
            out[((size_t)batch * TT + qi) * HH + n * 16 + ncol] = o[n][vv] * inv;
    }
}

// ---------------------------------------------------------------------------
extern "C" void kernel_launch(void* const* d_in, const int* in_sizes, int n_in,
                              void* d_out, int out_size, void* d_ws, size_t ws_size,
                              hipStream_t stream) {
    const float* x  = (const float*)d_in[0];
    const float* Wq = (const float*)d_in[1];
    const float* Wk = (const float*)d_in[2];
    const float* Wv = (const float*)d_in[3];

    // Workspace layout (bf16 elements)
    __bf16* xb = (__bf16*)d_ws;                       // [BT, CC]      16 MB
    __bf16* wt = xb + (size_t)BT * CC;                // [3, HH, CC]  384 KB
    __bf16* qw = wt + (size_t)3 * HH * CC;            // [BT, HH]       1 MB
    __bf16* kw = qw + (size_t)BT * HH;                // [BT, HH]       1 MB
    __bf16* vw = kw + (size_t)BT * HH;                // [BB, HH, TT]   1 MB

    cvt_x_kernel<<<(size_t)BT * CC / 4 / 256, 256, 0, stream>>>(x, xb);
    cvt_w_kernel<<<3 * HH * CC / 256, 256, 0, stream>>>(Wq, Wk, Wv, wt);
    proj_kernel<<<dim3(BT / 16, 3), 128, 0, stream>>>(xb, wt, qw, kw, vw);
    attn_kernel<<<dim3(TT / 64, BB), 128, 0, stream>>>(qw, kw, vw, (float*)d_out);
}
